// ChimeraV2Layer_9234179686604
// MI455X (gfx1250) — matrix operand order repaired and checked by reading, not compiled, not run on hardware
//
#include <hip/hip_runtime.h>

// ---------------- problem constants ----------------
constexpr int Bc    = 4;
constexpr int Sc    = 2048;     // power of two
constexpr int SSH   = 11;       // log2(Sc)
constexpr int Dc    = 1024;
constexpr int DHh   = 32;       // slr head dim
constexpr int NSLOT = 64;
constexpr int NBUS  = 128;
constexpr int ROWS  = Bc * Sc;  // 8192 token rows
#define EPSf 1.1920929e-07f

// ---------------- WMMA fragment types (CDNA5, wave32) ----------------
typedef __attribute__((ext_vector_type(16))) _Float16 h16;
typedef __attribute__((ext_vector_type(8)))  _Float16 h8;
typedef __attribute__((ext_vector_type(8)))  float    f8;

__device__ inline f8 f8zero() {
  f8 z;
#pragma unroll
  for (int i = 0; i < 8; ++i) z[i] = 0.f;
  return z;
}

__device__ inline h16 hcombine(h8 a, h8 b) {
  return __builtin_shufflevector(a, b, 0, 1, 2, 3, 4, 5, 6, 7,
                                 8, 9, 10, 11, 12, 13, 14, 15);
}

// A-matrix 16x32 f16 fragment (row-major source).
// Lane needs row (lane&15), K = 8*hi+[0..7] and 16+8*hi+[0..7]:
// two contiguous 16-byte chunks -> two b128 loads.
__device__ inline h16 load_a_frag(const _Float16* A, int lda) {
  const int lane = threadIdx.x & 31;
  const int row  = lane & 15;
  const int hi   = lane >> 4;
  const _Float16* r = A + (size_t)row * lda + (hi << 3);
  h8 a = *(const h8*)(r);
  h8 b = *(const h8*)(r + 16);
  return hcombine(a, b);
}

// B-matrix 32x16 f16 fragment from a TRANSPOSED (n-major) source:
// element (k,n) lives at Bt[n*ldn + k]. Lane n needs K = 8*hi+[0..7] and 16+8*hi+[0..7]:
// two contiguous 16-byte chunks -> two b128 loads.
__device__ inline h16 load_b_frag_t(const _Float16* Bt, int ldn) {
  const int lane = threadIdx.x & 31;
  const int n    = lane & 15;
  const int hi   = lane >> 4;
  const _Float16* r = Bt + (size_t)n * ldn + (hi << 3);
  h8 a = *(const h8*)(r);
  h8 b = *(const h8*)(r + 16);
  return hcombine(a, b);
}

// C/D 16x16 f32: VGPR i, lane<16 -> (M=i, N=lane); lane>=16 -> (M=8+i, N=lane-16)
__device__ inline void store_c_f32(float* C, int ldc, const f8& c) {
  const int lane = threadIdx.x & 31;
  const int n    = lane & 15;
  const int m0   = (lane >> 4) << 3;
#pragma unroll
  for (int i = 0; i < 8; ++i) C[(size_t)(m0 + i) * ldc + n] = c[i];
}

__device__ inline f8 wmma16(const h16& a, const h16& b, const f8& c) {
  return __builtin_amdgcn_wmma_f32_16x16x32_f16(
      /*neg_a=*/false, a, /*neg_b=*/false, b,
      /*c_mod=*/(short)0, c, /*reuse_a=*/false, /*reuse_b=*/false);
}

// ---------------- utility converts ----------------
__global__ __launch_bounds__(256) void k_f32_to_f16(const float* __restrict__ s,
                                                    _Float16* __restrict__ d, int n) {
  int i = blockIdx.x * 256 + threadIdx.x;
  if (i < n) d[i] = (_Float16)s[i];
}

// transpose-convert: src is [K][N] f32 row-major, dst is [N][K] f16 (n-major)
// K,N are powers of two here -> div/mod lower to shifts.
__global__ __launch_bounds__(256) void k_convT(const float* __restrict__ src,
                                               _Float16* __restrict__ dst,
                                               int K, int N) {
  int i = blockIdx.x * 256 + threadIdx.x;
  if (i < K * N) {
    int k = i / N, n = i % N;
    dst[(size_t)n * K + k] = (_Float16)src[i];
  }
}

// ---------------- 1) RMS norm -> xn (f16) ----------------
__global__ __launch_bounds__(256) void k_rmsnorm(const float* __restrict__ x,
                                                 const float* __restrict__ w,
                                                 _Float16* __restrict__ xn) {
  const int row = blockIdx.x;
  const float* xr = x + (size_t)row * Dc;
  float vbuf[4];
  float ss = 0.f;
#pragma unroll
  for (int i = 0; i < 4; ++i) {
    float v = xr[threadIdx.x + i * 256];
    vbuf[i] = v;
    ss += v * v;
  }
#pragma unroll
  for (int m = 16; m; m >>= 1) ss += __shfl_xor(ss, m, 32);
  __shared__ float red[8];
  if ((threadIdx.x & 31) == 0) red[threadIdx.x >> 5] = ss;
  __syncthreads();
  float tot = 0.f;
#pragma unroll
  for (int i = 0; i < 8; ++i) tot += red[i];
  const float rs = rsqrtf(tot * (1.f / Dc) + EPSf);
#pragma unroll
  for (int i = 0; i < 4; ++i) {
    int d = threadIdx.x + i * 256;
    xn[(size_t)row * Dc + d] = (_Float16)(vbuf[i] * rs * w[d]);
  }
}

// ---------------- 2) route_in = mean_s xn ----------------
__global__ __launch_bounds__(256) void k_route_mean(const _Float16* __restrict__ xn,
                                                    float* __restrict__ route_in) {
  const int b = blockIdx.y;
  const int d = blockIdx.x * 256 + threadIdx.x;
  const _Float16* p = xn + (size_t)b * Sc * Dc + d;
  float s = 0.f;
  for (int t = 0; t < Sc; ++t) s += (float)p[(size_t)t * Dc];
  route_in[b * Dc + d] = s * (1.f / Sc);
}

// ---------------- 3) router MLP -> coeff = p_med+p_full, pfull ----------------
__global__ __launch_bounds__(256) void k_router(const float* __restrict__ route_in,
                                                const float* __restrict__ w1,
                                                const float* __restrict__ w2,
                                                const float* __restrict__ b2,
                                                float* __restrict__ coeff,
                                                float* __restrict__ pfull) {
  const int b = blockIdx.x;
  __shared__ float h[512];
  const float* ri = route_in + b * Dc;
  for (int j = threadIdx.x; j < 512; j += 256) {
    float a = 0.f;
    for (int d = 0; d < Dc; ++d) a += ri[d] * w1[(size_t)d * 512 + j];
    h[j] = a / (1.f + __expf(-a));               // silu
  }
  __syncthreads();
  if (threadIdx.x == 0) {
    float lg[3];
#pragma unroll
    for (int c = 0; c < 3; ++c) {
      float a = b2[c];
      for (int j = 0; j < 512; ++j) a += h[j] * w2[j * 3 + c];
      lg[c] = a;
    }
    float mx = fmaxf(lg[0], fmaxf(lg[1], lg[2]));
    float e0 = __expf(lg[0] - mx), e1 = __expf(lg[1] - mx), e2 = __expf(lg[2] - mx);
    float inv = 1.f / (e0 + e1 + e2);
    coeff[b] = (e1 + e2) * inv;                  // p_medium + p_full
    pfull[b] = e2 * inv;
  }
}

// ---------------- 4) projections: (8192 x 1024) @ (1024 x 32) via WMMA --------
// WT is the TRANSPOSED weight [32][1024] (n-major).
// transpose_out=1 stores result as [b][dh][s] (n-major for later B-fragments).
__global__ __launch_bounds__(32) void k_proj(const _Float16* __restrict__ xn,
                                             const _Float16* __restrict__ WT,
                                             _Float16* __restrict__ out,
                                             int transpose_out) {
  const int mt = blockIdx.x;                     // row tile
  const int nt = blockIdx.y;                     // 0..1 (N=32)
  f8 acc = f8zero();
  const _Float16* Arow = xn + (size_t)mt * 16 * Dc;
  const _Float16* Bt   = WT + (size_t)(nt * 16) * Dc;
  for (int k = 0; k < Dc; k += 32) {
    h16 a = load_a_frag(Arow + k, Dc);
    h16 b = load_b_frag_t(Bt + k, Dc);
    acc = wmma16(a, b, acc);
  }
  const int lane = threadIdx.x & 31;
  const int n    = (lane & 15) + nt * 16;
  const int m0   = (lane >> 4) << 3;
  if (!transpose_out) {
#pragma unroll
    for (int i = 0; i < 8; ++i)
      out[(size_t)(mt * 16 + m0 + i) * DHh + n] = (_Float16)acc[i];
  } else {
#pragma unroll
    for (int i = 0; i < 8; ++i) {
      const int row = mt * 16 + m0 + i;
      const int b = row >> SSH, s = row & (Sc - 1);
      out[((size_t)b * DHh + n) * Sc + s] = (_Float16)acc[i];
    }
  }
}

// ---------------- 5) arc context for last token (tiny) ----------------
__global__ __launch_bounds__(256) void k_arc(const _Float16* __restrict__ xn,
                                             const float* __restrict__ lm,
                                             float* __restrict__ arc) {
  const int b = blockIdx.x;
  __shared__ float w[NSLOT];
  __shared__ float wn[NSLOT];
  const _Float16* q = xn + (size_t)(b * Sc + Sc - 1) * Dc;
  if (threadIdx.x < NSLOT) {
    float a = 0.f;
    const float* l = lm + (size_t)threadIdx.x * Dc;
    for (int d = 0; d < Dc; ++d) a += (float)q[d] * l[d];
    w[threadIdx.x] = a * (1.f / 32.f);           // 1/sqrt(1024)
  }
  __syncthreads();
  if (threadIdx.x == 0) {
    float mx = -1e30f;
    for (int k = 0; k < NSLOT; ++k) mx = fmaxf(mx, w[k]);
    float s = 0.f;
    for (int k = 0; k < NSLOT; ++k) { wn[k] = __expf(w[k] - mx); s += wn[k]; }
    float inv = 1.f / s;
    for (int k = 0; k < NSLOT; ++k) wn[k] *= inv;
  }
  __syncthreads();
  for (int d = threadIdx.x; d < Dc; d += 256) {
    float a = 0.f;
    for (int k = 0; k < NSLOT; ++k) a += wn[k] * lm[(size_t)k * Dc + d];
    arc[b * Dc + d] = a;
  }
}

// ---------------- 6) dual flash attention, one wave per 16-query tile --------
// k1/k2 are row-major (s,dh): that IS the transposed operand for QK^T.
// vT is [b][dh][s]: transposed operand for PV.
__global__ __launch_bounds__(32) void k_attn(const _Float16* __restrict__ q1,
                                             const _Float16* __restrict__ q2,
                                             const _Float16* __restrict__ k1,
                                             const _Float16* __restrict__ k2,
                                             const _Float16* __restrict__ vT,
                                             const float* __restrict__ lam_p,
                                             _Float16* __restrict__ ctx) {
  const int tile = blockIdx.x;                   // 0..ROWS/16
  const int row0 = tile * 16;
  const int b    = row0 >> SSH;
  const float lam = lam_p[0];
  const int lane = threadIdx.x & 31;
  const int nn   = lane & 15;
  const int half = lane >> 4;

  const h16 aq1 = load_a_frag(q1 + (size_t)row0 * DHh, DHh);
  const h16 aq2 = load_a_frag(q2 + (size_t)row0 * DHh, DHh);

  f8 o1a = f8zero(), o1b = f8zero(), o2a = f8zero(), o2b = f8zero();
  float m1[8], l1[8], m2[8], l2[8];
#pragma unroll
  for (int i = 0; i < 8; ++i) { m1[i] = -1e30f; m2[i] = -1e30f; l1[i] = 0.f; l2[i] = 0.f; }

  __shared__ _Float16 p1s[16 * 32];
  __shared__ _Float16 p2s[16 * 32];
  const float sc = 0.17677669529663687f;         // 1/sqrt(32)

  const _Float16* k1b = k1 + (size_t)b * Sc * DHh;
  const _Float16* k2b = k2 + (size_t)b * Sc * DHh;
  const _Float16* vtb = vT + (size_t)b * DHh * Sc;

  for (int k0 = 0; k0 < Sc; k0 += 32) {
    h16 bk;
    bk = load_b_frag_t(k1b + (size_t)k0 * DHh, DHh);        f8 s1a = wmma16(aq1, bk, f8zero());
    bk = load_b_frag_t(k1b + (size_t)(k0 + 16) * DHh, DHh); f8 s1b = wmma16(aq1, bk, f8zero());
    bk = load_b_frag_t(k2b + (size_t)k0 * DHh, DHh);        f8 s2a = wmma16(aq2, bk, f8zero());
    bk = load_b_frag_t(k2b + (size_t)(k0 + 16) * DHh, DHh); f8 s2b = wmma16(aq2, bk, f8zero());

    // online softmax per map; row stats replicated across the 16 lanes of a half
#pragma unroll
    for (int i = 0; i < 8; ++i) {
      const int m = half * 8 + i;
      {
        float sa = s1a[i] * sc, sb = s1b[i] * sc;
        float tm = fmaxf(sa, sb);
#pragma unroll
        for (int msk = 1; msk < 16; msk <<= 1) tm = fmaxf(tm, __shfl_xor(tm, msk, 32));
        float mn   = fmaxf(m1[i], tm);
        float corr = __expf(m1[i] - mn);
        float pa = __expf(sa - mn), pb = __expf(sb - mn);
        float ts = pa + pb;
#pragma unroll
        for (int msk = 1; msk < 16; msk <<= 1) ts += __shfl_xor(ts, msk, 32);
        l1[i] = l1[i] * corr + ts;
        o1a[i] *= corr; o1b[i] *= corr;
        m1[i] = mn;
        p1s[m * 32 + nn]      = (_Float16)pa;
        p1s[m * 32 + 16 + nn] = (_Float16)pb;
      }
      {
        float sa = s2a[i] * sc, sb = s2b[i] * sc;
        float tm = fmaxf(sa, sb);
#pragma unroll
        for (int msk = 1; msk < 16; msk <<= 1) tm = fmaxf(tm, __shfl_xor(tm, msk, 32));
        float mn   = fmaxf(m2[i], tm);
        float corr = __expf(m2[i] - mn);
        float pa = __expf(sa - mn), pb = __expf(sb - mn);
        float ts = pa + pb;
#pragma unroll
        for (int msk = 1; msk < 16; msk <<= 1) ts += __shfl_xor(ts, msk, 32);
        l2[i] = l2[i] * corr + ts;
        o2a[i] *= corr; o2b[i] *= corr;
        m2[i] = mn;
        p2s[m * 32 + nn]      = (_Float16)pa;
        p2s[m * 32 + 16 + nn] = (_Float16)pb;
      }
    }
    __syncthreads();

    const h16 ap1 = load_a_frag(p1s, 32);
    const h16 ap2 = load_a_frag(p2s, 32);
    const h16 bv0 = load_b_frag_t(vtb + k0, Sc);                    // dh 0..15
    const h16 bv1 = load_b_frag_t(vtb + (size_t)16 * Sc + k0, Sc);  // dh 16..31
    o1a = wmma16(ap1, bv0, o1a);
    o1b = wmma16(ap1, bv1, o1b);
    o2a = wmma16(ap2, bv0, o2a);
    o2b = wmma16(ap2, bv1, o2b);
    __syncthreads();
  }

#pragma unroll
  for (int i = 0; i < 8; ++i) {
    const float i1 = 1.f / l1[i], i2 = 1.f / l2[i];
    const int m = half * 8 + i;
    const size_t r = (size_t)(row0 + m) * DHh;
    ctx[r + nn]      = (_Float16)(o1a[i] * i1 - lam * o2a[i] * i2);
    ctx[r + 16 + nn] = (_Float16)(o1b[i] * i1 - lam * o2b[i] * i2);
  }
}

// ---------------- 7) EMA scan (4096 independent chains) ----------------
__global__ __launch_bounds__(256) void k_ssm(const _Float16* __restrict__ xn,
                                             const float* __restrict__ lambdas,
                                             float* __restrict__ ssm) {
  const int gid = blockIdx.x * 256 + threadIdx.x;  // 0..4095
  const int dh = gid & 127;
  const int h  = (gid >> 7) & 7;
  const int b  = gid >> 10;
  const float lam = lambdas[h];
  const float oml = 1.f - lam;
  const size_t base = (size_t)b * Sc * Dc + h * 128 + dh;
  float st = 0.f;
  for (int s = 0; s < Sc; ++s) {
    const size_t idx = base + (size_t)s * Dc;
    st = lam * st + oml * (float)xn[idx];
    ssm[idx] = st;
  }
}

// ---------------- 8) slr_out = ctx @ wo, fused with mix ----------------
// woT is [1024][32] (n-major).
__global__ __launch_bounds__(32) void k_slr_fuse(const _Float16* __restrict__ ctx,
                                                 const _Float16* __restrict__ woT,
                                                 const float* __restrict__ ssm,
                                                 const float* __restrict__ coeff,
                                                 const float* __restrict__ pfull,
                                                 const float* __restrict__ arc,
                                                 _Float16* __restrict__ mixed) {
  const int mt = blockIdx.x;                     // ROWS/16
  const int nt = blockIdx.y;                     // Dc/16
  const h16 a  = load_a_frag(ctx + (size_t)mt * 16 * DHh, DHh);  // K=32 in one step
  const h16 bw = load_b_frag_t(woT + (size_t)(nt * 16) * DHh, DHh);
  f8 acc = wmma16(a, bw, f8zero());
  const int lane = threadIdx.x & 31;
  const int nn   = (lane & 15) + nt * 16;
  const int m0   = (lane >> 4) << 3;
#pragma unroll
  for (int i = 0; i < 8; ++i) {
    const int row = mt * 16 + m0 + i;
    const int b = row >> SSH, s = row & (Sc - 1);
    float val = ssm[(size_t)row * Dc + nn] + coeff[b] * acc[i];
    if (s == Sc - 1) val += pfull[b] * arc[b * Dc + nn];
    mixed[(size_t)row * Dc + nn] = (_Float16)val;
  }
}

// ---------------- 9) big GEMM: [mixed | xn] (8192x2048) @ mix_w ----------------
// mwT is the transposed mix weight: [1024][2048] (n-major).
// 256 threads = 8 waves; block tile 64(M) x 64(N); waves in 4x2; each wave does
// two 16x16 accumulators. A and B staged in LDS with b128 vector ld/st.
__global__ __launch_bounds__(256) void k_mix_gemm(const _Float16* __restrict__ mixed,
                                                  const _Float16* __restrict__ xn,
                                                  const _Float16* __restrict__ mwT,
                                                  float* __restrict__ outp) {
  __shared__ _Float16 As[64][32];    // row-major (m,k)
  __shared__ _Float16 Bs[64][32];    // n-major  (n,k)
  const int mt = blockIdx.x;         // 0..127
  const int nt = blockIdx.y;         // 0..15
  const int wave = threadIdx.x >> 5;
  const int mw = wave >> 1;          // 0..3
  const int nw = wave & 1;           // 0..1
  const int t  = threadIdx.x;
  const int lr = t >> 2;             // 0..63 (row for staging)
  const int lc = (t & 3) << 3;       // chunk of 8 f16

  f8 acc0 = f8zero(), acc1 = f8zero();

  for (int half = 0; half < 2; ++half) {
    const _Float16* Asrc = half ? xn : mixed;
    for (int k0 = 0; k0 < Dc; k0 += 32) {
      __builtin_prefetch(Asrc + (size_t)(mt * 64 + lr) * Dc + k0 + 256, 0, 1);
      *(h8*)&As[lr][lc] =
          *(const h8*)(Asrc + (size_t)(mt * 64 + lr) * Dc + k0 + lc);
      *(h8*)&Bs[lr][lc] =
          *(const h8*)(mwT + (size_t)(nt * 64 + lr) * (2 * Dc) + half * Dc + k0 + lc);
      __syncthreads();
      h16 a  = load_a_frag(&As[mw * 16][0], 32);
      h16 b0 = load_b_frag_t(&Bs[nw * 32][0], 32);
      h16 b1 = load_b_frag_t(&Bs[nw * 32 + 16][0], 32);
      acc0 = wmma16(a, b0, acc0);
      acc1 = wmma16(a, b1, acc1);
      __syncthreads();
    }
  }
  float* Cp = outp + (size_t)(mt * 64 + mw * 16) * Dc + nt * 64 + nw * 32;
  store_c_f32(Cp, Dc, acc0);
  store_c_f32(Cp + 16, Dc, acc1);
}

// ---------------- 10) out = residual + rms(out_pre) ----------------
__global__ __launch_bounds__(256) void k_rms_out(const float* __restrict__ outp,
                                                 const float* __restrict__ x,
                                                 const float* __restrict__ w,
                                                 float* __restrict__ out) {
  const int row = blockIdx.x;
  const float* r = outp + (size_t)row * Dc;
  float vbuf[4];
  float ss = 0.f;
#pragma unroll
  for (int i = 0; i < 4; ++i) {
    float v = r[threadIdx.x + i * 256];
    vbuf[i] = v;
    ss += v * v;
  }
#pragma unroll
  for (int m = 16; m; m >>= 1) ss += __shfl_xor(ss, m, 32);
  __shared__ float red[8];
  if ((threadIdx.x & 31) == 0) red[threadIdx.x >> 5] = ss;
  __syncthreads();
  float tot = 0.f;
#pragma unroll
  for (int i = 0; i < 8; ++i) tot += red[i];
  const float rs = rsqrtf(tot * (1.f / Dc) + EPSf);
#pragma unroll
  for (int i = 0; i < 4; ++i) {
    const int d = threadIdx.x + i * 256;
    out[(size_t)row * Dc + d] = x[(size_t)row * Dc + d] + vbuf[i] * rs * w[d];
  }
}

// ---------------- 11) mean over S of final out ----------------
__global__ __launch_bounds__(256) void k_out_mean(const float* __restrict__ out,
                                                  float* __restrict__ mean) {
  const int b = blockIdx.y;
  const int d = blockIdx.x * 256 + threadIdx.x;
  const float* p = out + (size_t)b * Sc * Dc + d;
  float s = 0.f;
  for (int t = 0; t < Sc; ++t) s += p[(size_t)t * Dc];
  mean[b * Dc + d] = s * (1.f / Sc);
}

// ---------------- 12) bus: summary->publish->(trivial softmax)->expand --------
__global__ __launch_bounds__(128) void k_bus(const float* __restrict__ mean,
                                             const float* __restrict__ wc,
                                             const float* __restrict__ wp,
                                             const float* __restrict__ we,
                                             float* __restrict__ bus_vec) {
  const int b = blockIdx.x;
  __shared__ float summ[NBUS], pub[NBUS];
  const int j = threadIdx.x;
  {
    float a = 0.f;
    for (int d = 0; d < Dc; ++d) a += mean[b * Dc + d] * wc[(size_t)d * NBUS + j];
    summ[j] = a;
  }
  __syncthreads();
  {
    float a = 0.f;
    for (int k = 0; k < NBUS; ++k) a += summ[k] * wp[k * NBUS + j];
    pub[j] = a;                                  // softmax over 1 key == 1
  }
  __syncthreads();
  for (int d = threadIdx.x; d < Dc; d += 128) {
    float a = 0.f;
    for (int k = 0; k < NBUS; ++k) a += pub[k] * we[(size_t)k * Dc + d];
    bus_vec[b * Dc + d] = 0.1f * a;
  }
}

// ---------------- 13) broadcast-add bus vector ----------------
__global__ __launch_bounds__(256) void k_bus_add(float* __restrict__ out,
                                                 const float* __restrict__ bus_vec) {
  const size_t i = (size_t)blockIdx.x * 256 + threadIdx.x;
  const int d = (int)(i & (Dc - 1));
  const int b = (int)(i >> (SSH + 10));          // / (Sc*Dc)
  out[i] += bus_vec[b * Dc + d];
}

// ======================= host launch =======================
extern "C" void kernel_launch(void* const* d_in, const int* in_sizes, int n_in,
                              void* d_out, int out_size, void* d_ws, size_t ws_size,
                              hipStream_t stream) {
  (void)in_sizes; (void)n_in; (void)out_size; (void)ws_size;

  const float* x    = (const float*)d_in[0];
  const float* wni  = (const float*)d_in[1];
  const float* wno  = (const float*)d_in[2];
  const float* rw1  = (const float*)d_in[3];
  const float* rw2  = (const float*)d_in[4];
  const float* rb2  = (const float*)d_in[5];
  // d_in[6..9] = ttt_* : dead code in the reference (ttt_scale never used)
  const float* lambdas = (const float*)d_in[10];
  const float* swq1 = (const float*)d_in[11];
  const float* swq2 = (const float*)d_in[12];
  const float* swk1 = (const float*)d_in[13];
  const float* swk2 = (const float*)d_in[14];
  const float* swv  = (const float*)d_in[15];
  const float* swo  = (const float*)d_in[16];
  const float* slam = (const float*)d_in[17];
  const float* lm   = (const float*)d_in[18];
  const float* mixw = (const float*)d_in[19];
  const float* busc = (const float*)d_in[20];
  const float* buse = (const float*)d_in[21];
  const float* busp = (const float*)d_in[22];
  float* out = (float*)d_out;

  // -------- workspace carve-out --------
  char* w = (char*)d_ws;
  auto alloc = [&](size_t bytes) {
    char* p = w;
    w += (bytes + 255) & ~(size_t)255;
    return p;
  };
  _Float16* xn    = (_Float16*)alloc((size_t)ROWS * Dc * 2);
  _Float16* mixed = (_Float16*)alloc((size_t)ROWS * Dc * 2);
  float*    ssm   = (float*)alloc((size_t)ROWS * Dc * 4);
  float*    outp  = (float*)alloc((size_t)ROWS * Dc * 4);
  _Float16* q1  = (_Float16*)alloc((size_t)ROWS * DHh * 2);
  _Float16* q2  = (_Float16*)alloc((size_t)ROWS * DHh * 2);
  _Float16* k1  = (_Float16*)alloc((size_t)ROWS * DHh * 2);
  _Float16* k2  = (_Float16*)alloc((size_t)ROWS * DHh * 2);
  _Float16* vT  = (_Float16*)alloc((size_t)ROWS * DHh * 2);
  _Float16* ctx = (_Float16*)alloc((size_t)ROWS * DHh * 2);
  _Float16* wq1T = (_Float16*)alloc((size_t)Dc * DHh * 2);
  _Float16* wq2T = (_Float16*)alloc((size_t)Dc * DHh * 2);
  _Float16* wk1T = (_Float16*)alloc((size_t)Dc * DHh * 2);
  _Float16* wk2T = (_Float16*)alloc((size_t)Dc * DHh * 2);
  _Float16* wvT  = (_Float16*)alloc((size_t)Dc * DHh * 2);
  _Float16* woT  = (_Float16*)alloc((size_t)DHh * Dc * 2);
  _Float16* mwT  = (_Float16*)alloc((size_t)2 * Dc * Dc * 2);
  float* route_in = (float*)alloc((size_t)Bc * Dc * 4);
  float* coeff    = (float*)alloc(Bc * 4);
  float* pfull    = (float*)alloc(Bc * 4);
  float* arc      = (float*)alloc((size_t)Bc * Dc * 4);
  float* meanv    = (float*)alloc((size_t)Bc * Dc * 4);
  float* busv     = (float*)alloc((size_t)Bc * Dc * 4);

  // -------- weight transpose-conversions to f16 (n-major) --------
  const int nProj = Dc * DHh;
  k_convT<<<(nProj + 255) / 256, 256, 0, stream>>>(swq1, wq1T, Dc, DHh);
  k_convT<<<(nProj + 255) / 256, 256, 0, stream>>>(swq2, wq2T, Dc, DHh);
  k_convT<<<(nProj + 255) / 256, 256, 0, stream>>>(swk1, wk1T, Dc, DHh);
  k_convT<<<(nProj + 255) / 256, 256, 0, stream>>>(swk2, wk2T, Dc, DHh);
  k_convT<<<(nProj + 255) / 256, 256, 0, stream>>>(swv, wvT, Dc, DHh);
  k_convT<<<(nProj + 255) / 256, 256, 0, stream>>>(swo, woT, DHh, Dc);
  const int nMW = 2 * Dc * Dc;
  k_convT<<<(nMW + 255) / 256, 256, 0, stream>>>(mixw, mwT, 2 * Dc, Dc);

  // -------- pipeline --------
  k_rmsnorm<<<ROWS, 256, 0, stream>>>(x, wni, xn);
  k_route_mean<<<dim3(Dc / 256, Bc), 256, 0, stream>>>(xn, route_in);
  k_router<<<Bc, 256, 0, stream>>>(route_in, rw1, rw2, rb2, coeff, pfull);

  k_proj<<<dim3(ROWS / 16, 2), 32, 0, stream>>>(xn, wq1T, q1, 0);
  k_proj<<<dim3(ROWS / 16, 2), 32, 0, stream>>>(xn, wq2T, q2, 0);
  k_proj<<<dim3(ROWS / 16, 2), 32, 0, stream>>>(xn, wk1T, k1, 0);
  k_proj<<<dim3(ROWS / 16, 2), 32, 0, stream>>>(xn, wk2T, k2, 0);
  k_proj<<<dim3(ROWS / 16, 2), 32, 0, stream>>>(xn, wvT, vT, 1);

  k_arc<<<Bc, 256, 0, stream>>>(xn, lm, arc);
  k_attn<<<ROWS / 16, 32, 0, stream>>>(q1, q2, k1, k2, vT, slam, ctx);
  k_ssm<<<16, 256, 0, stream>>>(xn, lambdas, ssm);
  k_slr_fuse<<<dim3(ROWS / 16, Dc / 16), 32, 0, stream>>>(ctx, woT, ssm, coeff, pfull, arc, mixed);

  k_mix_gemm<<<dim3(ROWS / 64, Dc / 64), 256, 0, stream>>>(mixed, xn, mwT, outp);

  k_rms_out<<<ROWS, 256, 0, stream>>>(outp, x, wno, out);
  k_out_mean<<<dim3(Dc / 256, Bc), 256, 0, stream>>>(out, meanv);
  k_bus<<<Bc, 128, 0, stream>>>(meanv, busc, busp, buse, busv);
  k_bus_add<<<(int)((size_t)ROWS * Dc / 256), 256, 0, stream>>>(out, busv);
}
